// MultiHeadAttention_53730040873768
// MI455X (gfx1250) — matrix operand order repaired and checked by reading, not compiled, not run on hardware
//
#include <hip/hip_runtime.h>
#include <hip/hip_bf16.h>
#include <cstdint>
#include <cstddef>

// ---------------------------------------------------------------------------
// MI455X (gfx1250) MHA forward.
// bf16 WMMA (v_wmma_f32_16x16x32_bf16) with f32 accumulate, Tensor Data Mover
// (tensor_load_to_lds, TENSORcnt) double-buffered global->LDS staging, wave32
// flash attention with in-register online softmax.
// ---------------------------------------------------------------------------

typedef __attribute__((ext_vector_type(16))) __bf16        v16bf;
typedef __attribute__((ext_vector_type(8)))  float         v8f;
typedef __attribute__((ext_vector_type(4)))  unsigned int  v4u;
typedef __attribute__((ext_vector_type(8)))  int           v8i;
typedef __attribute__((ext_vector_type(4)))  int           v4i;

#define DIM   1024
#define NH    16
#define HD    64
#define BATCH 2
#define SEQ   2048

__device__ __forceinline__ unsigned short f2bf(float f) {
    unsigned u = __builtin_bit_cast(unsigned, f);
    u += 0x7FFFu + ((u >> 16) & 1u);          // round-to-nearest-even
    return (unsigned short)(u >> 16);
}

__device__ __forceinline__ v8f vzero8() {
    v8f z;
    for (int i = 0; i < 8; ++i) z[i] = 0.0f;
    return z;
}

// ---------------------------------------------------------------------------
// Tensor Data Mover: 2D tile (tile_x elems contiguous, tile_y rows, row pitch
// stride_elems) from global bf16 -> LDS. Issued once per wave; TENSORcnt.
// D# layout per CDNA5 ISA 8.3/8.4 (group0 128b, group1 256b; groups 2/3 zero
// for <=2D tiles). data_size=1 (2-byte elements). 6-arg builtin form
// (clang-23 / therock-10.0): (v4u, v8i, v4i, v4i, v8i, i32 cpol).
// ---------------------------------------------------------------------------
__device__ __forceinline__ void tdm_load_2d(unsigned lds_addr,
                                            const unsigned short* gptr,
                                            unsigned tile_x, unsigned tile_y,
                                            unsigned tensor_d0, unsigned tensor_d1,
                                            unsigned long long stride_elems) {
    unsigned long long ga = (unsigned long long)(uintptr_t)gptr;
    v4u g0;
    g0[0] = 1u;                                               // count=1 (user D#)
    g0[1] = lds_addr;                                         // lds_addr [63:32]
    g0[2] = (unsigned)(ga & 0xFFFFFFFFu);                     // global_addr lo
    g0[3] = (unsigned)((ga >> 32) & 0x01FFFFFFu) | (2u << 30);// addr hi | type=2
    v8i g1;
    g1[0] = (int)(1u << 16);                                  // data_size=1 (2B)
    g1[1] = (int)((tensor_d0 & 0xFFFFu) << 16);               // tensor_dim0 lo16
    g1[2] = (int)(((tensor_d0 >> 16) & 0xFFFFu) |
                  ((tensor_d1 & 0xFFFFu) << 16));             // d0 hi16 | d1 lo16
    g1[3] = (int)(((tensor_d1 >> 16) & 0xFFFFu) |
                  ((tile_x & 0xFFFFu) << 16));                // d1 hi16 | tile_dim0
    g1[4] = (int)(tile_y & 0xFFFFu);                          // tile_dim1 (dim2=0)
    g1[5] = (int)(stride_elems & 0xFFFFFFFFull);              // dim0_stride lo32
    g1[6] = (int)((stride_elems >> 32) & 0xFFFFull);          // dim0_stride hi16
    g1[7] = 0;
    v4i z4 = {0, 0, 0, 0};
    v8i z8 = {0, 0, 0, 0, 0, 0, 0, 0};
    __builtin_amdgcn_tensor_load_to_lds(g0, g1, z4, z4, z8, 0);
}

__device__ __forceinline__ unsigned lds_off(const void* p) {
    return (unsigned)(uintptr_t)p;    // LDS aperture: addr[31:0] = LDS offset
}

// ---------------------------------------------------------------------------
// fp32 -> bf16 bulk convert (n multiple of 4)
// ---------------------------------------------------------------------------
__global__ __launch_bounds__(256) void cvt_f32_bf16(const float* __restrict__ in,
                                                    unsigned short* __restrict__ out,
                                                    int n) {
    int i = (blockIdx.x * 256 + threadIdx.x) * 4;
    if (i + 3 < n) {
        float4 v = *reinterpret_cast<const float4*>(in + i);
        ushort4 o;
        o.x = f2bf(v.x); o.y = f2bf(v.y); o.z = f2bf(v.z); o.w = f2bf(v.w);
        *reinterpret_cast<ushort4*>(out + i) = o;
    }
}

// ---------------------------------------------------------------------------
// bf16 GEMM:  C[M,N] = A[M,K] * W[N,K]^T   (torch Linear: y = x @ W.T)
// 256 threads = 8 waves; block tile 128x128; wave tile 64x32.
// TDM double-buffered LDS staging: wave0 issues next K-slab DMA, all waves
// compute on the current one.
// OUT_MODE 0: bf16 scattered to [B,H,S,D]; OUT_MODE 1: f32 row-major [M,N].
// ---------------------------------------------------------------------------
template <int OUT_MODE>
__global__ __launch_bounds__(256) void gemm_bf16(const unsigned short* __restrict__ A,
                                                 const unsigned short* __restrict__ W,
                                                 void* __restrict__ Out,
                                                 int M, int N, int K) {
    __shared__ __align__(32) unsigned short As[2][128 * 32];
    __shared__ __align__(32) unsigned short Bs[2][128 * 32];

    const int t    = threadIdx.x;
    const int lane = t & 31;
    const int wave = t >> 5;           // 0..7
    const int wm   = wave & 1;         // 64-row slab
    const int wn   = wave >> 1;        // 32-col slab
    const int m0   = blockIdx.x * 128;
    const int n0   = blockIdx.y * 128;

    v8f acc[4][2];
    for (int mi = 0; mi < 4; ++mi)
        for (int ni = 0; ni < 2; ++ni)
            acc[mi][ni] = vzero8();

    const int kHalf = (lane >> 4) * 16;

    // prologue: stage K-slab 0 via TDM (wave0 only; EXEC-independent DMA)
    if (t < 32) {
        tdm_load_2d(lds_off(&As[0][0]), A + (size_t)m0 * K, 32, 128,
                    (unsigned)K, (unsigned)M, (unsigned long long)K);
        tdm_load_2d(lds_off(&Bs[0][0]), W + (size_t)n0 * K, 32, 128,
                    (unsigned)K, (unsigned)N, (unsigned long long)K);
    }

    for (int k0 = 0, it = 0; k0 < K; k0 += 32, ++it) {
        const int buf = it & 1;
        if (t < 32) __builtin_amdgcn_s_wait_tensorcnt(0);  // slab `it` landed
        __syncthreads();
        if (t < 32 && k0 + 32 < K) {                       // prefetch slab it+1
            tdm_load_2d(lds_off(&As[buf ^ 1][0]), A + (size_t)m0 * K + k0 + 32,
                        32, 128, (unsigned)K, (unsigned)M, (unsigned long long)K);
            tdm_load_2d(lds_off(&Bs[buf ^ 1][0]), W + (size_t)n0 * K + k0 + 32,
                        32, 128, (unsigned)K, (unsigned)N, (unsigned long long)K);
        }

        v16bf bfrag[2];
        for (int ni = 0; ni < 2; ++ni) {
            int row = wn * 32 + ni * 16 + (lane & 15);
            bfrag[ni] = *reinterpret_cast<const v16bf*>(&Bs[buf][row * 32 + kHalf]);
        }
        for (int mi = 0; mi < 4; ++mi) {
            int row = wm * 64 + mi * 16 + (lane & 15);
            v16bf afrag = *reinterpret_cast<const v16bf*>(&As[buf][row * 32 + kHalf]);
            for (int ni = 0; ni < 2; ++ni) {
                acc[mi][ni] = __builtin_amdgcn_wmma_f32_16x16x32_bf16(
                    false, afrag, false, bfrag[ni], (short)0, acc[mi][ni], false, false);
            }
        }
        __syncthreads();   // all waves done with buf before slab it+2 overwrites
    }

    // epilogue: WMMA C layout -> (m, n). vgpr r, lanes<16 => M=r, lanes>=16 => M=r+8
    const int halfAdd = (lane < 16) ? 0 : 8;
    const int nLane   = lane & 15;
    for (int mi = 0; mi < 4; ++mi) {
        for (int ni = 0; ni < 2; ++ni) {
            for (int r = 0; r < 8; ++r) {
                int m = m0 + wm * 64 + mi * 16 + r + halfAdd;
                int n = n0 + wn * 32 + ni * 16 + nLane;
                float val = acc[mi][ni][r];
                if (OUT_MODE == 0) {
                    int b = m >> 11;        // m = b*SEQ + s
                    int s = m & (SEQ - 1);
                    int h = n >> 6;         // n = h*HD + d
                    int d = n & (HD - 1);
                    ((unsigned short*)Out)[(((size_t)b * NH + h) * SEQ + s) * HD + d] = f2bf(val);
                } else {
                    ((float*)Out)[(size_t)m * N + n] = val;
                }
            }
        }
    }
}

// ---------------------------------------------------------------------------
// Flash attention (causal). Grid: (SEQ/64, BATCH*NH). 128 threads = 4 waves.
// Q/K/V tiles are contiguous 64x64 bf16 slabs in [B,H,S,D] -> flat TDM copies,
// double-buffered across the key loop. Online softmax in registers via
// 16-lane shfl reductions; P round-trips through LDS to the A-operand layout.
// ---------------------------------------------------------------------------
__global__ __launch_bounds__(128) void attn_fwd(const unsigned short* __restrict__ Xq,
                                                const unsigned short* __restrict__ Xk,
                                                const unsigned short* __restrict__ Xv,
                                                unsigned short* __restrict__ Out) {
    __shared__ __align__(32) unsigned short Qs[64 * 64];
    __shared__ __align__(32) unsigned short Ks[2][64 * 64];
    __shared__ __align__(32) unsigned short Vs[2][64 * 64];
    __shared__ __align__(32) unsigned short Ps[64 * 64];

    const int t    = threadIdx.x;
    const int lane = t & 31;
    const int wave = t >> 5;                 // 0..3 -> q rows wave*16..+15
    const int qt   = blockIdx.x;             // query tile of 64 rows
    const int bh   = blockIdx.y;             // b*NH + h
    const size_t headBase = (size_t)bh * SEQ * HD;
    const unsigned TILE = 64 * 64;           // contiguous bf16 elements per tile

    // prologue: TDM Q tile + K/V tile 0
    if (t < 32) {
        tdm_load_2d(lds_off(&Qs[0]),    Xq + headBase + (size_t)qt * TILE, TILE, 1,
                    SEQ * HD, 1, TILE);
        tdm_load_2d(lds_off(&Ks[0][0]), Xk + headBase,                     TILE, 1,
                    SEQ * HD, 1, TILE);
        tdm_load_2d(lds_off(&Vs[0][0]), Xv + headBase,                     TILE, 1,
                    SEQ * HD, 1, TILE);
    }

    const int kHalf = (lane >> 4) * 16;
    v16bf qfrag[2] = {};

    v8f acc_o[4];
    for (int dt = 0; dt < 4; ++dt) acc_o[dt] = vzero8();
    float m_i[8], l_i[8];
    for (int r = 0; r < 8; ++r) { m_i[r] = -1e30f; l_i[r] = 0.0f; }

    const int   halfSel = (lane < 16) ? 0 : 8;
    const int   nLane   = lane & 15;
    const float scale   = 0.125f;            // 1/sqrt(64)

    for (int kt = 0; kt <= qt; ++kt) {
        const int buf = kt & 1;
        if (t < 32) __builtin_amdgcn_s_wait_tensorcnt(0);    // tiles for kt landed
        __syncthreads();
        if (t < 32 && kt + 1 <= qt) {                        // prefetch kt+1
            tdm_load_2d(lds_off(&Ks[buf ^ 1][0]),
                        Xk + headBase + (size_t)(kt + 1) * TILE, TILE, 1,
                        SEQ * HD, 1, TILE);
            tdm_load_2d(lds_off(&Vs[buf ^ 1][0]),
                        Xv + headBase + (size_t)(kt + 1) * TILE, TILE, 1,
                        SEQ * HD, 1, TILE);
        }
        if (kt == 0) {   // Q resident now; build A fragments once
            for (int kc = 0; kc < 2; ++kc) {
                int row = wave * 16 + (lane & 15);
                qfrag[kc] = *reinterpret_cast<const v16bf*>(&Qs[row * 64 + kc * 32 + kHalf]);
            }
        }

        // ---- S = Q K^T (wave computes 16x64; 8 WMMA) ----
        v8f acc_s[4];
        for (int nt = 0; nt < 4; ++nt) acc_s[nt] = vzero8();
        for (int nt = 0; nt < 4; ++nt) {
            for (int kc = 0; kc < 2; ++kc) {
                int row = nt * 16 + (lane & 15);
                v16bf kf = *reinterpret_cast<const v16bf*>(&Ks[buf][row * 64 + kc * 32 + kHalf]);
                acc_s[nt] = __builtin_amdgcn_wmma_f32_16x16x32_bf16(
                    false, qfrag[kc], false, kf, (short)0, acc_s[nt], false, false);
            }
        }

        // ---- scale + causal mask (diagonal tile only) ----
        const bool diag = (kt == qt);
        for (int nt = 0; nt < 4; ++nt) {
            for (int r = 0; r < 8; ++r) {
                float s = acc_s[nt][r] * scale;
                if (diag) {
                    int qRow = wave * 16 + r + halfSel;
                    int kCol = nt * 16 + nLane;
                    if (kCol > qRow) s = -1e30f;
                }
                acc_s[nt][r] = s;
            }
        }

        // ---- online softmax: row m lives in one VGPR across one 16-lane half
        for (int r = 0; r < 8; ++r) {
            float mcur = acc_s[0][r];
            for (int nt = 1; nt < 4; ++nt) mcur = fmaxf(mcur, acc_s[nt][r]);
            for (int off = 8; off >= 1; off >>= 1)
                mcur = fmaxf(mcur, __shfl_xor(mcur, off, 16));
            float mnew = fmaxf(m_i[r], mcur);
            float psum = 0.0f;
            for (int nt = 0; nt < 4; ++nt) {
                float p = __expf(acc_s[nt][r] - mnew);
                acc_s[nt][r] = p;
                psum += p;
            }
            for (int off = 8; off >= 1; off >>= 1)
                psum += __shfl_xor(psum, off, 16);
            float alpha = __expf(m_i[r] - mnew);
            l_i[r] = l_i[r] * alpha + psum;
            m_i[r] = mnew;
            for (int dt = 0; dt < 4; ++dt) acc_o[dt][r] *= alpha;
        }

        // ---- P (bf16) -> LDS so it can re-enter as the WMMA A operand ----
        for (int nt = 0; nt < 4; ++nt) {
            for (int r = 0; r < 8; ++r) {
                int row = wave * 16 + r + halfSel;
                int col = nt * 16 + nLane;
                Ps[row * 64 + col] = f2bf(acc_s[nt][r]);
            }
        }
        __syncthreads();

        // ---- O += P @ V (8 WMMA per wave) ----
        for (int kc = 0; kc < 2; ++kc) {
            int prow = wave * 16 + (lane & 15);
            v16bf pf = *reinterpret_cast<const v16bf*>(&Ps[prow * 64 + kc * 32 + kHalf]);
            for (int dt = 0; dt < 4; ++dt) {
                v16bf vf;
                for (int s = 0; s < 16; ++s)
                    vf[s] = __builtin_bit_cast(__bf16, Vs[buf][(kc * 32 + kHalf + s) * 64 + dt * 16 + (lane & 15)]);
                acc_o[dt] = __builtin_amdgcn_wmma_f32_16x16x32_bf16(
                    false, pf, false, vf, (short)0, acc_o[dt], false, false);
            }
        }
        __syncthreads();   // done with buf before kt+2's TDM overwrites it
    }

    // ---- normalize and store bf16 back to [B, S, DIM] ----
    const int b = bh >> 4, h = bh & 15;
    for (int dt = 0; dt < 4; ++dt) {
        for (int r = 0; r < 8; ++r) {
            int row = qt * 64 + wave * 16 + r + halfSel;
            int col = h * HD + dt * 16 + nLane;
            float val = acc_o[dt][r] / l_i[r];
            Out[((size_t)b * SEQ + row) * DIM + col] = f2bf(val);
        }
    }
}

// ---------------------------------------------------------------------------
// Host-side orchestration. Workspace layout (bf16 = ushort), 64 MB total:
//  [0,8)    q bf16        [8,16)  k bf16       [16,24) v bf16
//  [24,26)  wq            [26,28) wk           [28,30) wv      [30,32) wo
//  [32,40)  xq [B,H,S,D]  [40,48) xk           [48,56) xv
//  [56,64)  attn out bf16 [B,S,DIM]
// ---------------------------------------------------------------------------
extern "C" void kernel_launch(void* const* d_in, const int* in_sizes, int n_in,
                              void* d_out, int out_size, void* d_ws, size_t ws_size,
                              hipStream_t stream) {
    const float* q  = (const float*)d_in[0];
    const float* k  = (const float*)d_in[1];
    const float* v  = (const float*)d_in[2];
    const float* wq = (const float*)d_in[3];
    const float* wk = (const float*)d_in[4];
    const float* wv = (const float*)d_in[5];
    const float* wo = (const float*)d_in[6];

    char* ws = (char*)d_ws;
    const size_t MB = 1ull << 20;
    unsigned short* qb  = (unsigned short*)(ws + 0 * MB);
    unsigned short* kb  = (unsigned short*)(ws + 8 * MB);
    unsigned short* vb  = (unsigned short*)(ws + 16 * MB);
    unsigned short* wqb = (unsigned short*)(ws + 24 * MB);
    unsigned short* wkb = (unsigned short*)(ws + 26 * MB);
    unsigned short* wvb = (unsigned short*)(ws + 28 * MB);
    unsigned short* wob = (unsigned short*)(ws + 30 * MB);
    unsigned short* xq  = (unsigned short*)(ws + 32 * MB);
    unsigned short* xk  = (unsigned short*)(ws + 40 * MB);
    unsigned short* xv  = (unsigned short*)(ws + 48 * MB);
    unsigned short* at  = (unsigned short*)(ws + 56 * MB);

    const int NTOK = BATCH * SEQ;       // 4096
    const int NACT = NTOK * DIM;        // 4,194,304
    const int NWT  = DIM * DIM;         // 1,048,576

    cvt_f32_bf16<<<NACT / 1024, 256, 0, stream>>>(q,  qb,  NACT);
    cvt_f32_bf16<<<NACT / 1024, 256, 0, stream>>>(k,  kb,  NACT);
    cvt_f32_bf16<<<NACT / 1024, 256, 0, stream>>>(v,  vb,  NACT);
    cvt_f32_bf16<<<NWT  / 1024, 256, 0, stream>>>(wq, wqb, NWT);
    cvt_f32_bf16<<<NWT  / 1024, 256, 0, stream>>>(wk, wkb, NWT);
    cvt_f32_bf16<<<NWT  / 1024, 256, 0, stream>>>(wv, wvb, NWT);
    cvt_f32_bf16<<<NWT  / 1024, 256, 0, stream>>>(wo, wob, NWT);

    dim3 gg(NTOK / 128, DIM / 128);     // 32 x 8
    gemm_bf16<0><<<gg, 256, 0, stream>>>(qb, wqb, xq, NTOK, DIM, DIM);
    gemm_bf16<0><<<gg, 256, 0, stream>>>(kb, wkb, xk, NTOK, DIM, DIM);
    gemm_bf16<0><<<gg, 256, 0, stream>>>(vb, wvb, xv, NTOK, DIM, DIM);

    dim3 ga(SEQ / 64, BATCH * NH);      // 32 x 32
    attn_fwd<<<ga, 128, 0, stream>>>(xq, xk, xv, at);

    gemm_bf16<1><<<gg, 256, 0, stream>>>(at, wob, d_out, NTOK, DIM, DIM);
}
